// PatchFeatureEmbedding_16312285791017
// MI455X (gfx1250) — matrix operand order repaired and checked by reading, not compiled, not run on hardware
//
#include <hip/hip_runtime.h>
#include <math.h>

typedef _Float16 v16h __attribute__((ext_vector_type(16)));
typedef float    v8f  __attribute__((ext_vector_type(8)));

#define T_LEN   6000
#define NPAT    199
#define PATCH   60
#define STRIDE  30
#define NPERSEG 30
#define SEG_STEP 15
#define NSEG    3
#define NF      16
#define KPAD    32
#define E1      128
#define E2      256
#define PI_F    3.14159265358979323846f

// workspace layout (bytes)
#define WS_TRIG   0                       // cos[16*30], sin[16*30], hann[30] floats
#define WS_W1T    4096                    // 128*32 f16  = 8 KB
#define WS_W2T    (4096 + 8192)           // 256*128 f16 = 64 KB
#define WS_FEAT   (4096 + 8192 + 65536)   // Mp*32 f16

// ---------------------------------------------------------------- prep ----
__global__ void pfe_prep(const float* __restrict__ W1, const float* __restrict__ W2,
                         float* __restrict__ trig, _Float16* __restrict__ w1t,
                         _Float16* __restrict__ w2t) {
  int t = threadIdx.x;
  for (int i = t; i < NF * NPERSEG; i += 256) {
    int f = i / NPERSEG, j = i % NPERSEG;
    float w = 2.0f * PI_F * (float)(f * j) / (float)NPERSEG;
    trig[i] = cosf(w);
    trig[NF * NPERSEG + i] = sinf(w);
  }
  for (int j = t; j < NPERSEG; j += 256)
    trig[2 * NF * NPERSEG + j] = 0.5f * (1.0f - cosf(2.0f * PI_F * (float)j / (float)NPERSEG));
  // W1T[n][k] (n-major), K padded 22->32
  for (int i = t; i < E1 * KPAD; i += 256) {
    int n = i / KPAD, k = i % KPAD;
    w1t[i] = (k < 22) ? (_Float16)W1[k * E1 + n] : (_Float16)0.0f;
  }
  // W2T[n][k] (n-major)
  for (int i = t; i < E2 * E1; i += 256) {
    int n = i / E1, k = i % E1;
    w2t[i] = (_Float16)W2[k * E2 + n];
  }
}

// ------------------------------------------------------------ features ----
__global__ void pfe_features(const float* __restrict__ x,
                             const float* __restrict__ g_m, const float* __restrict__ b_m,
                             const float* __restrict__ g_s, const float* __restrict__ b_s,
                             const float* __restrict__ trig,
                             _Float16* __restrict__ feat, int M, int Mp, int B) {
  int m = blockIdx.x * blockDim.x + threadIdx.x;
  if (m >= Mp) return;
  _Float16* row = feat + (size_t)m * KPAD;
  if (m >= M) {
    #pragma unroll
    for (int i = 0; i < KPAD; i++) row[i] = (_Float16)0.0f;
    return;
  }
  int b = m / NPAT, pi = m % NPAT;
  const float* px = x + (size_t)b * T_LEN + (size_t)pi * STRIDE;
  if (b + 1 < B) __builtin_prefetch(px + T_LEN, 0, 1);  // global_prefetch_b8

  float p[PATCH];
  #pragma unroll
  for (int j = 0; j < PATCH; j++) p[j] = px[j];

  // -------- morphological features
  float mx = p[0], mn = p[0], sum = 0.0f;
  #pragma unroll
  for (int j = 0; j < PATCH; j++) { mx = fmaxf(mx, p[j]); mn = fminf(mn, p[j]); sum += p[j]; }
  float mean = sum * (1.0f / PATCH);
  float m2 = 0.0f, m3 = 0.0f, m4 = 0.0f, en = 0.0f;
  #pragma unroll
  for (int j = 0; j < PATCH; j++) {
    float c = p[j] - mean, c2 = c * c;
    m2 += c2; m3 += c2 * c; m4 += c2 * c2; en += p[j] * p[j];
  }
  m2 *= (1.0f / PATCH); m3 *= (1.0f / PATCH); m4 *= (1.0f / PATCH);
  float stdv = sqrtf(m2);
  float bestA = -1.0f; int bestI = 0;
  #pragma unroll
  for (int j = 0; j < PATCH; j++) { float a = fabsf(p[j]); if (a > bestA) { bestA = a; bestI = j; } }
  float peak_loc = (float)bestI * (1.0f / PATCH);
  float dmx = p[1] - p[0], dmn = dmx, dab = 0.0f;
  int zc = 0;
  #pragma unroll
  for (int j = 0; j < PATCH - 1; j++) {
    float d = p[j + 1] - p[j];
    dmx = fmaxf(dmx, d); dmn = fminf(dmn, d); dab += fabsf(d);
    float a0 = p[j] + 1e-10f, a1 = p[j + 1] + 1e-10f;
    int s0 = (a0 > 0.0f) - (a0 < 0.0f), s1 = (a1 > 0.0f) - (a1 < 0.0f);
    zc += (s0 != s1);
  }
  dab *= (1.0f / (PATCH - 1));
  float kurt = m4 / (m2 * m2) - 3.0f;
  float skew = m3 / (m2 * stdv);
  float morph[13] = { mx, mn, mx - mn, mean, stdv, peak_loc,
                      dmx, dmn, dab, (float)zc * (1.0f / PATCH), en, kurt, skew };
  #pragma unroll
  for (int i = 0; i < 13; i++) if (!__builtin_isfinite(morph[i])) morph[i] = 0.0f;

  // -------- Welch PSD (detrend const, periodic hann, density, one-sided)
  const float* cosT = trig;
  const float* sinT = trig + NF * NPERSEG;
  const float* hann = trig + 2 * NF * NPERSEG;
  float wn = 0.0f;
  #pragma unroll
  for (int j = 0; j < NPERSEG; j++) wn += hann[j] * hann[j];
  wn *= 200.0f;  // SFREQ * sum(w^2)
  float psd[NF];
  #pragma unroll
  for (int f = 0; f < NF; f++) psd[f] = 0.0f;
  for (int s = 0; s < NSEG; s++) {
    int base = s * SEG_STEP;
    float mu = 0.0f;
    #pragma unroll
    for (int j = 0; j < NPERSEG; j++) mu += p[base + j];
    mu *= (1.0f / NPERSEG);
    float d[NPERSEG];
    #pragma unroll
    for (int j = 0; j < NPERSEG; j++) d[j] = (p[base + j] - mu) * hann[j];
    for (int f = 0; f < NF; f++) {
      float re = 0.0f, im = 0.0f;
      #pragma unroll
      for (int j = 0; j < NPERSEG; j++) {
        re += d[j] * cosT[f * NPERSEG + j];
        im += d[j] * sinT[f * NPERSEG + j];
      }
      psd[f] += re * re + im * im;
    }
  }
  #pragma unroll
  for (int f = 0; f < NF; f++)
    psd[f] *= ((f == 0 || f == NF - 1) ? 1.0f : 2.0f) * (1.0f / (NSEG * 1.0f)) / wn;

  float total = 0.0f;
  #pragma unroll
  for (int f = 0; f < NF; f++) total += psd[f];
  total += 1e-12f;
  float inv_total = 1.0f / total;
  float bandp[5] = {0, 0, 0, 0, 0};
  float bestP = psd[0]; int bestF = 0;
  float cs = 0.0f; int edgeI = -1;
  float ent = 0.0f;
  #pragma unroll
  for (int f = 0; f < NF; f++) {
    float fr = (float)f * (200.0f / 30.0f);
    if (fr >= 1.0f  && fr <= 4.0f)  bandp[0] += psd[f];
    if (fr >= 4.0f  && fr <= 8.0f)  bandp[1] += psd[f];
    if (fr >= 8.0f  && fr <= 13.0f) bandp[2] += psd[f];
    if (fr >= 13.0f && fr <= 30.0f) bandp[3] += psd[f];
    if (fr >= 30.0f && fr <= 99.0f) bandp[4] += psd[f];
    if (psd[f] > bestP) { bestP = psd[f]; bestF = f; }
    cs += psd[f];
    if (edgeI < 0 && cs >= 0.95f * total) edgeI = f;
    float pn = psd[f] * inv_total;
    ent -= pn * log2f(pn + 1e-12f);
  }
  if (edgeI < 0) edgeI = 0;
  float spec[9] = { bandp[0] * inv_total, bandp[1] * inv_total, bandp[2] * inv_total,
                    bandp[3] * inv_total, bandp[4] * inv_total,
                    (float)bestF * (200.0f / 30.0f), (float)edgeI * (200.0f / 30.0f),
                    ent, total };
  #pragma unroll
  for (int i = 0; i < 9; i++) if (!__builtin_isfinite(spec[i])) spec[i] = 0.0f;

  // -------- per-patch layernorms, write f16 K-padded row
  {
    float mu = 0.0f;
    #pragma unroll
    for (int i = 0; i < 13; i++) mu += morph[i];
    mu *= (1.0f / 13.0f);
    float va = 0.0f;
    #pragma unroll
    for (int i = 0; i < 13; i++) { float d = morph[i] - mu; va += d * d; }
    float rs = rsqrtf(va * (1.0f / 13.0f) + 1e-5f);
    #pragma unroll
    for (int i = 0; i < 13; i++) row[i] = (_Float16)((morph[i] - mu) * rs * g_m[i] + b_m[i]);
  }
  {
    float mu = 0.0f;
    #pragma unroll
    for (int i = 0; i < 9; i++) mu += spec[i];
    mu *= (1.0f / 9.0f);
    float va = 0.0f;
    #pragma unroll
    for (int i = 0; i < 9; i++) { float d = spec[i] - mu; va += d * d; }
    float rs = rsqrtf(va * (1.0f / 9.0f) + 1e-5f);
    #pragma unroll
    for (int i = 0; i < 9; i++) row[13 + i] = (_Float16)((spec[i] - mu) * rs * g_s[i] + b_s[i]);
  }
  #pragma unroll
  for (int i = 22; i < KPAD; i++) row[i] = (_Float16)0.0f;
}

// ----------------------------------------------------------------- mlp ----
// 8 waves / block, each wave owns a 16-row M tile. WMMA f32_16x16x32_f16.
__global__ __launch_bounds__(256, 1)
void pfe_mlp(const _Float16* __restrict__ feat, const _Float16* __restrict__ w1t,
             const _Float16* __restrict__ w2t,
             const float* __restrict__ b1, const float* __restrict__ b2,
             const float* __restrict__ g_o, const float* __restrict__ b_o,
             float* __restrict__ out, int M, int Mp) {
  __shared__ _Float16 hst[8][16][E1 + 8];  // +8 halves row pad: conflict-free TR reads

  int lane = threadIdx.x & 31;
  int wv   = threadIdx.x >> 5;
  int gm   = blockIdx.x * 8 + wv;
  int m0   = gm * 16;
  if (m0 >= Mp) return;                    // wave-uniform; EXEC all-ones for WMMA
  int  l16  = lane & 15;
  bool hi   = (lane & 16) != 0;
  int  kA   = hi ? 16 : 0;                 // A-frag K half (contiguous, matches B)
  int  kB   = hi ? 16 : 0;                 // B-frag K half (contiguous, matches A)

  // ---- A fragment for GEMM1 (row-major feat, K padded to 32)
  const _Float16* fr = feat + (size_t)(m0 + l16) * KPAD;
  v16h a1;
  #pragma unroll
  for (int i = 0; i < 8; i++) { a1[i] = fr[kA + i]; a1[8 + i] = fr[kA + 8 + i]; }

  // ---- GEMM1 (16x128), bias + exact GELU, restage f16 into LDS
  #pragma unroll
  for (int nt = 0; nt < 8; nt++) {
    int col = nt * 16 + l16;
    const _Float16* wr = w1t + col * KPAD + kB;
    v16h bf;
    #pragma unroll
    for (int i = 0; i < 16; i++) bf[i] = wr[i];
    v8f c = {};
    c = __builtin_amdgcn_wmma_f32_16x16x32_f16(false, a1, false, bf, (short)0, c, false, false);
    float bias = b1[col];
    #pragma unroll
    for (int r = 0; r < 8; r++) {
      float v = c[r] + bias;
      v = 0.5f * v * (1.0f + erff(v * 0.70710678118654752f));  // exact gelu
      hst[wv][r + (hi ? 8 : 0)][col] = (_Float16)v;
    }
  }

  // ---- A fragments for GEMM2 (K=128 -> 4 chunks of 32), from LDS
  v16h af[4];
  #pragma unroll
  for (int kt = 0; kt < 4; kt++) {
    #pragma unroll
    for (int i = 0; i < 8; i++) {
      af[kt][i]     = hst[wv][l16][kt * 32 + kA + i];
      af[kt][8 + i] = hst[wv][l16][kt * 32 + kA + 8 + i];
    }
  }

  // ---- GEMM2 (16x256), + b2, keep all 16 C tiles in registers for rowwise LN
  float cc[16][8];
  #pragma unroll
  for (int nt = 0; nt < 16; nt++) {
    int col = nt * 16 + l16;
    v8f c = {};
    #pragma unroll
    for (int kt = 0; kt < 4; kt++) {
      const _Float16* wr = w2t + col * E1 + kt * 32 + kB;
      v16h bf;
      #pragma unroll
      for (int i = 0; i < 16; i++) bf[i] = wr[i];
      c = __builtin_amdgcn_wmma_f32_16x16x32_f16(false, af[kt], false, bf, (short)0, c, false, false);
    }
    float bias = b2[col];
    #pragma unroll
    for (int r = 0; r < 8; r++) cc[nt][r] = c[r] + bias;
  }

  // ---- rowwise LayerNorm: a row lives across one 16-lane half (N) x 16 tiles
  float s[8], s2[8];
  #pragma unroll
  for (int r = 0; r < 8; r++) {
    s[r] = 0.0f; s2[r] = 0.0f;
    #pragma unroll
    for (int nt = 0; nt < 16; nt++) { float v = cc[nt][r]; s[r] += v; s2[r] += v * v; }
  }
  #pragma unroll
  for (int d = 1; d < 16; d <<= 1) {
    #pragma unroll
    for (int r = 0; r < 8; r++) {
      s[r]  += __shfl_xor(s[r],  d, 16);
      s2[r] += __shfl_xor(s2[r], d, 16);
    }
  }
  #pragma unroll
  for (int r = 0; r < 8; r++) {
    float mean = s[r] * (1.0f / E2);
    float var  = s2[r] * (1.0f / E2) - mean * mean;
    float rstd = rsqrtf(var + 1e-5f);
    int row = m0 + r + (hi ? 8 : 0);
    if (row < M) {
      #pragma unroll
      for (int nt = 0; nt < 16; nt++) {
        int col = nt * 16 + l16;
        out[(size_t)row * E2 + col] = (cc[nt][r] - mean) * rstd * g_o[col] + b_o[col];
      }
    }
  }
}

// --------------------------------------------------------------- launch ----
extern "C" void kernel_launch(void* const* d_in, const int* in_sizes, int n_in,
                              void* d_out, int out_size, void* d_ws, size_t ws_size,
                              hipStream_t stream) {
  (void)n_in; (void)out_size; (void)ws_size;
  const float* x   = (const float*)d_in[0];
  const float* g_m = (const float*)d_in[1];
  const float* b_m = (const float*)d_in[2];
  const float* g_s = (const float*)d_in[3];
  const float* b_s = (const float*)d_in[4];
  const float* W1  = (const float*)d_in[5];
  const float* b1  = (const float*)d_in[6];
  const float* W2  = (const float*)d_in[7];
  const float* b2  = (const float*)d_in[8];
  const float* g_o = (const float*)d_in[9];
  const float* b_o = (const float*)d_in[10];

  int B  = in_sizes[0] / T_LEN;
  int M  = B * NPAT;
  int Mp = (M + 15) & ~15;

  char* ws = (char*)d_ws;
  float*    trig = (float*)(ws + WS_TRIG);
  _Float16* w1t  = (_Float16*)(ws + WS_W1T);
  _Float16* w2t  = (_Float16*)(ws + WS_W2T);
  _Float16* feat = (_Float16*)(ws + WS_FEAT);

  pfe_prep<<<1, 256, 0, stream>>>(W1, W2, trig, w1t, w2t);
  pfe_features<<<(Mp + 255) / 256, 256, 0, stream>>>(x, g_m, b_m, g_s, b_s, trig, feat, M, Mp, B);
  int tiles = Mp / 16;                       // one wave per tile
  pfe_mlp<<<(tiles + 7) / 8, 256, 0, stream>>>(feat, w1t, w2t, b1, b2, g_o, b_o,
                                               (float*)d_out, M, Mp);
}
